// LstmNetMinVar_57372173140198
// MI455X (gfx1250) — matrix operand order repaired and checked by reading, not compiled, not run on hardware
//
#include <hip/hip_runtime.h>
#include <hip/hip_bf16.h>

// ---------------------------------------------------------------------------
// MI455X (gfx1250, wave32) implementation of LstmNetMinVar.
// GEMM stages (input+recurrent gates, covariance Gram, Q=cov*cov) use
// v_wmma_f32_16x16x32_f16 (f16 in / f32 accumulate), with 8-accumulator
// register blocking per wave. The gate GEMM stages its B operand into LDS
// with global_load_async_to_lds_b128 (ASYNCcnt) and reads fragments via
// ds_read_b128. Elementwise LSTM cell, instance-norm, and the FISTA QP
// solver run on f32 VALU.
// ---------------------------------------------------------------------------

typedef __attribute__((ext_vector_type(16))) _Float16 v16h;
typedef __attribute__((ext_vector_type(8)))  _Float16 v8h;
typedef __attribute__((ext_vector_type(8)))  float    v8f;

#define T_STEPS 256   // LSTM time steps (== S of the QP problems)
#define BATCH   512   // LSTM batch (== L of covariance)
#define ADIM    128   // hidden/input width

// ---- WMMA fragment loaders (contiguous k order in A and B, wave32) ------------------
// A (16x32, MxK): lane l: m=l&15, grp=l>>4; halves 0..15 = K: 16*grp ..
//                 16*grp+15 (contiguous).
static __device__ __forceinline__ v16h ld_a16(const _Float16* p) {
  v8h lo = *(const v8h*)(p);
  v8h hi = *(const v8h*)(p + 8);
  return __builtin_shufflevector(lo, hi, 0,1,2,3,4,5,6,7,8,9,10,11,12,13,14,15);
}
// B (32x16, KxN) with B stored transposed ([n][k] row-major): lane l: n=l&15,
// grp=l>>4; halves 0..15 = K: 16*grp .. 16*grp+15 (contiguous).
static __device__ __forceinline__ v16h ld_b16(const _Float16* p) {
  v8h lo = *(const v8h*)(p);
  v8h hi = *(const v8h*)(p + 8);
  return __builtin_shufflevector(lo, hi, 0,1,2,3,4,5,6,7,8,9,10,11,12,13,14,15);
}
static __device__ __forceinline__ v8f wmma_f16(v16h a, v16h b, v8f c) {
  return __builtin_amdgcn_wmma_f32_16x16x32_f16(false, a, false, b, (short)0, c,
                                                false, false);
}

static __device__ __forceinline__ float sigf(float x) {
  return 1.0f / (1.0f + expf(-x));
}

// ---- weight prep: Wcat^T[n][k] (n=512 gates, k=256 = [W_ih | W_hh]) --------
__global__ void prep_weights(const float* __restrict__ W_ih,
                             const float* __restrict__ W_hh,
                             const float* __restrict__ b_ih,
                             const float* __restrict__ b_hh,
                             _Float16* __restrict__ WcatT,
                             float* __restrict__ bias) {
  int n = blockIdx.x;          // 512
  int k = threadIdx.x;         // 256
  float v = (k < ADIM) ? W_ih[n * ADIM + k] : W_hh[n * ADIM + (k - ADIM)];
  WcatT[n * 256 + k] = (_Float16)v;
  if (k == 0) bias[n] = b_ih[n] + b_hh[n];
}

__global__ void init_state(_Float16* __restrict__ h16, float* __restrict__ c32) {
  int i = blockIdx.x * blockDim.x + threadIdx.x;   // 65536
  h16[i] = (_Float16)0.0f;
  c32[i] = 0.0f;
}

// ---- instance norm over (L,A)=65536 per s; emit f16 xn ---------------------
__global__ __launch_bounds__(256) void inst_norm(const float* __restrict__ x,
                                                 const float* __restrict__ in_w,
                                                 const float* __restrict__ in_b,
                                                 _Float16* __restrict__ xn16) {
  __shared__ float s1[256], s2[256];
  int s = blockIdx.x, tid = threadIdx.x;
  const float* xp = x + (size_t)s * 65536;
  float sum = 0.f, sq = 0.f;
  for (int i = tid * 4; i < 65536; i += 1024) {
    float4 v = *(const float4*)(xp + i);
    sum += v.x + v.y + v.z + v.w;
    sq  += v.x * v.x + v.y * v.y + v.z * v.z + v.w * v.w;
  }
  s1[tid] = sum; s2[tid] = sq; __syncthreads();
  for (int st = 128; st > 0; st >>= 1) {
    if (tid < st) { s1[tid] += s1[tid + st]; s2[tid] += s2[tid + st]; }
    __syncthreads();
  }
  float mean  = s1[0] * (1.0f / 65536.0f);
  float var   = s2[0] * (1.0f / 65536.0f) - mean * mean;
  float scale = rsqrtf(var + 1e-5f) * in_w[0];
  float shift = in_b[0] - mean * scale;
  _Float16* op = xn16 + (size_t)s * 65536;
  for (int i = tid * 4; i < 65536; i += 1024) {
    float4 v = *(const float4*)(xp + i);
    op[i + 0] = (_Float16)(v.x * scale + shift);
    op[i + 1] = (_Float16)(v.y * scale + shift);
    op[i + 2] = (_Float16)(v.z * scale + shift);
    op[i + 3] = (_Float16)(v.w * scale + shift);
  }
}

// ---- per-step gate GEMM: gates(512x512) = [xn_t | h] (512x256) x Wcat ------
// B tile (128 gate columns x 256 K, 64KB) is staged to LDS with
// global_load_async_to_lds_b128; fragments then come from LDS (ds_read_b128).
__global__ __launch_bounds__(256) void gates_gemm(const _Float16* __restrict__ xn16,
                                                  const _Float16* __restrict__ h16,
                                                  const _Float16* __restrict__ WcatT,
                                                  float* __restrict__ gates, int t) {
  __shared__ _Float16 Bs[128 * 256];   // 64 KB

  int tid  = threadIdx.x;
  int lane = tid & 31, l15 = lane & 15, grp = lane >> 4;
  int wave = tid >> 5;                       // 8 waves, one M-tile row each
  int M0 = blockIdx.x * 128, N0 = blockIdx.y * 128;

  // ---- async stage of B tile: Wcat rows [N0, N0+128) (4096 x 16B chunks).
  // The dsaddr VGPR is derived from ptrtoint(&Bs): the low 32 bits of a
  // generic LDS pointer are the wave-relative LDS byte offset (aperture bits
  // live in [63:32]), and taking the address makes Bs escape into the
  // memory-clobbering asm, so the compiler must treat Bs as written here
  // (keeps the ds_read fragment loads below live and distinct).
  {
    unsigned bs_base = (unsigned)(size_t)(&Bs[0]);
    const _Float16* src = WcatT + (size_t)N0 * 256;
    for (int i = tid; i < 4096; i += 256) {
      unsigned lds_off = bs_base + (unsigned)i * 16u;
      unsigned long long ga =
          (unsigned long long)(size_t)src + (unsigned long long)i * 16ull;
      asm volatile("global_load_async_to_lds_b128 %0, %1, off"
                   :: "v"(lds_off), "v"(ga)
                   : "memory");
    }
    asm volatile("s_wait_asynccnt 0x0" ::: "memory");
    __syncthreads();
  }

  int row = M0 + 16 * wave + l15;            // A-operand row (batch index)
  const _Float16* xrow = xn16 + ((size_t)t * BATCH + row) * ADIM;
  const _Float16* hrow = h16 + (size_t)row * ADIM;

  v8f acc[8];
#pragma unroll
  for (int nt = 0; nt < 8; ++nt)
    acc[nt] = (v8f){0.f,0.f,0.f,0.f,0.f,0.f,0.f,0.f};

#pragma unroll
  for (int kk = 0; kk < 256; kk += 32) {
    const _Float16* pa = (kk < ADIM) ? (xrow + kk + 16 * grp)
                                     : (hrow + (kk - ADIM) + 16 * grp);
    v16h a = ld_a16(pa);          // one A fragment, reused for 8 WMMAs
#pragma unroll
    for (int nt = 0; nt < 8; ++nt) {
      const _Float16* pb = &Bs[(16 * nt + l15) * 256 + kk + 16 * grp];
      v16h b = ld_b16(pb);        // ds_read_b128 x2
      acc[nt] = wmma_f16(a, b, acc[nt]);
    }
  }

#pragma unroll
  for (int nt = 0; nt < 8; ++nt) {
    int col = N0 + 16 * nt + l15;
#pragma unroll
    for (int r = 0; r < 8; ++r)
      gates[(size_t)(M0 + 16 * wave + r + 8 * grp) * 512 + col] = acc[nt][r];
  }
}

// ---- LSTM cell elementwise; writes h (f16) and transposed history ----------
__global__ __launch_bounds__(128) void lstm_cell(const float* __restrict__ gates,
                                                 const float* __restrict__ bias,
                                                 float* __restrict__ c32,
                                                 _Float16* __restrict__ h16,
                                                 _Float16* __restrict__ hsT,
                                                 int t) {
  int b = blockIdx.x;         // 512
  int a = threadIdx.x;        // 128
  const float* gr = gates + (size_t)b * 512;
  float xi = gr[a]           + bias[a];
  float xf = gr[ADIM + a]    + bias[ADIM + a];
  float xg = gr[2*ADIM + a]  + bias[2*ADIM + a];
  float xo = gr[3*ADIM + a]  + bias[3*ADIM + a];
  float c  = c32[b * ADIM + a];
  float cn = sigf(xf) * c + sigf(xi) * tanhf(xg);
  float h  = sigf(xo) * tanhf(cn);
  c32[b * ADIM + a] = cn;
  h16[b * ADIM + a] = (_Float16)h;
  hsT[(size_t)t * 65536 + (size_t)a * BATCH + b] = (_Float16)h;   // [s][a][l]
}

// ---- per-s batch mean over l ----------------------------------------------
__global__ __launch_bounds__(128) void batch_mean(const _Float16* __restrict__ hsT,
                                                  float* __restrict__ mu) {
  int t = blockIdx.x, a = threadIdx.x;
  const _Float16* p = hsT + (size_t)t * 65536 + (size_t)a * BATCH;
  float s = 0.f;
  for (int b = 0; b < BATCH; ++b) s += (float)p[b];
  mu[t * ADIM + a] = s * (1.0f / (float)BATCH);
}

// ---- Gram matrix C = H^T H per s (M=N=128, K=512) via WMMA -----------------
__global__ __launch_bounds__(256) void cov_gemm(const _Float16* __restrict__ hsT,
                                                float* __restrict__ cov) {
  int s = blockIdx.x;
  int tid = threadIdx.x, lane = tid & 31, l15 = lane & 15, grp = lane >> 4;
  int wave = tid >> 5;
  const _Float16* base = hsT + (size_t)s * 65536;
  const _Float16* arow = base + (size_t)(16 * wave + l15) * BATCH;

  v8f acc[8];
#pragma unroll
  for (int nt = 0; nt < 8; ++nt)
    acc[nt] = (v8f){0.f,0.f,0.f,0.f,0.f,0.f,0.f,0.f};

#pragma unroll 2
  for (int kk = 0; kk < BATCH; kk += 32) {
    v16h a = ld_a16(arow + kk + 16 * grp);     // reused for 8 WMMAs
#pragma unroll
    for (int nt = 0; nt < 8; ++nt) {
      v16h b = ld_b16(base + (size_t)(16 * nt + l15) * BATCH + kk + 16 * grp);
      acc[nt] = wmma_f16(a, b, acc[nt]);
    }
  }

#pragma unroll
  for (int nt = 0; nt < 8; ++nt)
#pragma unroll
    for (int r = 0; r < 8; ++r)
      cov[(size_t)s * 16384 + (size_t)(16 * wave + r + 8 * grp) * ADIM +
          16 * nt + l15] = acc[nt][r];
}

// ---- cov = (Gram - L*mu*mu^T)/(L-1), halve off-diagonal; emit f16 ----------
__global__ __launch_bounds__(256) void cov_finalize(float* __restrict__ cov,
                                                    _Float16* __restrict__ cov16,
                                                    const float* __restrict__ mu) {
  size_t idx = (size_t)blockIdx.x * 256 + threadIdx.x;   // 256*16384 total
  int s = (int)(idx >> 14);
  int r = (int)(idx & 16383);
  int i = r >> 7, j = r & 127;
  float c  = cov[idx];
  float m_ = (c - (float)BATCH * mu[s * ADIM + i] * mu[s * ADIM + j]) *
             (1.0f / (float)(BATCH - 1));
  float val = (i == j) ? m_ : 0.5f * m_;
  cov[idx]   = val;
  cov16[idx] = (_Float16)val;
}

// ---- Q = cov * cov per s (symmetric cov -> both operands contiguous) -------
__global__ __launch_bounds__(256) void q_gemm(const _Float16* __restrict__ cov16,
                                              float* __restrict__ Q) {
  int s = blockIdx.x;
  int tid = threadIdx.x, lane = tid & 31, l15 = lane & 15, grp = lane >> 4;
  int wave = tid >> 5;
  const _Float16* base = cov16 + (size_t)s * 16384;
  const _Float16* arow = base + (size_t)(16 * wave + l15) * ADIM;

  v8f acc[8];
#pragma unroll
  for (int nt = 0; nt < 8; ++nt)
    acc[nt] = (v8f){0.f,0.f,0.f,0.f,0.f,0.f,0.f,0.f};

#pragma unroll
  for (int kk = 0; kk < ADIM; kk += 32) {
    v16h a = ld_a16(arow + kk + 16 * grp);
#pragma unroll
    for (int nt = 0; nt < 8; ++nt) {
      // symmetry: column n of cov == row n of cov (contiguous)
      v16h b = ld_b16(base + (size_t)(16 * nt + l15) * ADIM + kk + 16 * grp);
      acc[nt] = wmma_f16(a, b, acc[nt]);
    }
  }

#pragma unroll
  for (int nt = 0; nt < 8; ++nt)
#pragma unroll
    for (int r = 0; r < 8; ++r)
      Q[(size_t)s * 16384 + (size_t)(16 * wave + r + 8 * grp) * ADIM +
        16 * nt + l15] = acc[nt][r];
}

// ---- block reductions (blockDim == 128) ------------------------------------
static __device__ __forceinline__ float blk_sum(float v, float* red, int tid) {
  red[tid] = v; __syncthreads();
#pragma unroll
  for (int st = 64; st > 0; st >>= 1) {
    if (tid < st) red[tid] += red[tid + st];
    __syncthreads();
  }
  float r = red[0]; __syncthreads(); return r;
}
static __device__ __forceinline__ float blk_min(float v, float* red, int tid) {
  red[tid] = v; __syncthreads();
#pragma unroll
  for (int st = 64; st > 0; st >>= 1) {
    if (tid < st) red[tid] = fminf(red[tid], red[tid + st]);
    __syncthreads();
  }
  float r = red[0]; __syncthreads(); return r;
}
static __device__ __forceinline__ float blk_max(float v, float* red, int tid) {
  red[tid] = v; __syncthreads();
#pragma unroll
  for (int st = 64; st > 0; st >>= 1) {
    if (tid < st) red[tid] = fmaxf(red[tid], red[tid + st]);
    __syncthreads();
  }
  float r = red[0]; __syncthreads(); return r;
}

// ---- per-s QP: power iteration + FISTA with simplex-box projection ---------
// One workgroup per s; thread a owns Q row a in registers (unrolled dots),
// z broadcast from LDS.
__global__ __launch_bounds__(128) void solve_minvar(const float* __restrict__ Q,
                                                    float* __restrict__ out) {
  __shared__ float red[128];
  __shared__ float zv[128];
  int s = blockIdx.x, tid = threadIdx.x;
  float qrow[ADIM];
  const float* qp = Q + (size_t)s * 16384 + (size_t)tid * ADIM;
#pragma unroll
  for (int k = 0; k < ADIM; ++k) qrow[k] = qp[k];

  // power iteration (30 iters) for step size
  zv[tid] = 0.08838834764831845f;   // 1/sqrt(128)
  __syncthreads();
  for (int it = 0; it < 30; ++it) {
    float nb = 0.f;
#pragma unroll
    for (int k = 0; k < ADIM; ++k) nb += qrow[k] * zv[k];
    float nrm = blk_sum(nb * nb, red, tid);
    zv[tid] = nb / (sqrtf(nrm) + 1e-12f);
    __syncthreads();
  }
  float qb = 0.f;
#pragma unroll
  for (int k = 0; k < ADIM; ++k) qb += qrow[k] * zv[k];
  float lam  = blk_sum(zv[tid] * qb, red, tid);
  float step = 1.0f / (lam + 1e-8f);

  // FISTA (300 iters)
  float w = 1.0f / (float)ADIM, tfac = 1.0f;
  __syncthreads();
  zv[tid] = w;
  __syncthreads();
  for (int it = 0; it < 300; ++it) {
    float g = 0.f;
#pragma unroll
    for (int k = 0; k < ADIM; ++k) g += qrow[k] * zv[k];
    float v = zv[tid] - step * g;
    // project onto {sum(w)=1, -1<=w<=1}: bisection on tau + active-set refit
    float lo = blk_min(v, red, tid) - 1.0f;
    float hi = blk_max(v, red, tid) + 1.0f;
    for (int bi = 0; bi < 50; ++bi) {
      float mid  = 0.5f * (lo + hi);
      float cl   = fminf(fmaxf(v - mid, -1.0f), 1.0f);
      float ssum = blk_sum(cl, red, tid);
      if (ssum > 1.0f) lo = mid; else hi = mid;   // uniform branch
    }
    float tau0 = 0.5f * (lo + hi);
    float w0   = fminf(fmaxf(v - tau0, -1.0f), 1.0f);
    bool  inter = fabsf(w0) < 1.0f - 1e-6f;
    float kcnt = fmaxf(blk_sum(inter ? 1.0f : 0.0f, red, tid), 1.0f);
    float bsum = blk_sum(inter ? 0.0f : w0, red, tid);
    float vsum = blk_sum(inter ? v : 0.0f, red, tid);
    float tau  = (vsum - (1.0f - bsum)) / kcnt;
    float wn   = fminf(fmaxf(v - tau, -1.0f), 1.0f);
    float tn   = 0.5f * (1.0f + sqrtf(1.0f + 4.0f * tfac * tfac));
    float zn   = wn + (tfac - 1.0f) / tn * (wn - w);
    w = wn; tfac = tn;
    __syncthreads();
    zv[tid] = zn;
    __syncthreads();
  }
  out[(size_t)s * ADIM + tid] = w;
}

// ---------------------------------------------------------------------------
extern "C" void kernel_launch(void* const* d_in, const int* in_sizes, int n_in,
                              void* d_out, int out_size, void* d_ws, size_t ws_size,
                              hipStream_t stream) {
  const float* x    = (const float*)d_in[0];   // (256,1,512,128)
  const float* in_w = (const float*)d_in[1];
  const float* in_b = (const float*)d_in[2];
  const float* W_ih = (const float*)d_in[3];   // (512,128)
  const float* W_hh = (const float*)d_in[4];   // (512,128)
  const float* b_ih = (const float*)d_in[5];   // (512,)
  const float* b_hh = (const float*)d_in[6];   // (512,)
  float* out = (float*)d_out;                  // (256,128)

  char* ws = (char*)d_ws;
  size_t off = 0;
  auto take = [&](size_t bytes) {
    char* p = ws + off;
    off += (bytes + 255) & ~(size_t)255;
    return (void*)p;
  };
  _Float16* xn16  = (_Float16*)take((size_t)T_STEPS * BATCH * ADIM * 2); // 32MB
  _Float16* hsT   = (_Float16*)take((size_t)T_STEPS * BATCH * ADIM * 2); // 32MB
  float*    cov32 = (float*)   take((size_t)T_STEPS * ADIM * ADIM * 4);  // 16MB
  float*    Q32   = (float*)   take((size_t)T_STEPS * ADIM * ADIM * 4);  // 16MB
  _Float16* cov16 = (_Float16*)take((size_t)T_STEPS * ADIM * ADIM * 2);  //  8MB
  float*    gates = (float*)   take((size_t)BATCH * 512 * 4);            //  1MB
  _Float16* WcatT = (_Float16*)take((size_t)512 * 256 * 2);
  _Float16* h16   = (_Float16*)take((size_t)BATCH * ADIM * 2);
  float*    c32   = (float*)   take((size_t)BATCH * ADIM * 4);
  float*    bias  = (float*)   take((size_t)512 * 4);
  float*    mu    = (float*)   take((size_t)T_STEPS * ADIM * 4);

  prep_weights<<<512, 256, 0, stream>>>(W_ih, W_hh, b_ih, b_hh, WcatT, bias);
  init_state  <<<256, 256, 0, stream>>>(h16, c32);
  inst_norm   <<<256, 256, 0, stream>>>(x, in_w, in_b, xn16);

  for (int t = 0; t < T_STEPS; ++t) {
    gates_gemm<<<dim3(4, 4), 256, 0, stream>>>(xn16, h16, WcatT, gates, t);
    lstm_cell <<<512, 128, 0, stream>>>(gates, bias, c32, h16, hsT, t);
  }

  batch_mean  <<<256, 128, 0, stream>>>(hsT, mu);
  cov_gemm    <<<256, 256, 0, stream>>>(hsT, cov32);
  cov_finalize<<<16384, 256, 0, stream>>>(cov32, cov16, mu);
  q_gemm      <<<256, 256, 0, stream>>>(cov16, Q32);
  solve_minvar<<<256, 128, 0, stream>>>(Q32, out);
}
